// modelMol_30777735644030
// MI455X (gfx1250) — matrix operand order repaired and checked by reading, not compiled, not run on hardware
//
#include <hip/hip_runtime.h>
#include <hip/hip_bf16.h>

typedef __attribute__((ext_vector_type(16))) _Float16 v16h;
typedef __attribute__((ext_vector_type(8)))  _Float16 v8h;
typedef __attribute__((ext_vector_type(4)))  _Float16 v4h;
typedef __attribute__((ext_vector_type(8)))  float    v8f;

// ---------------------------------------------------------------------------
// Prep kernels: convert fp32 inputs/weights to f16 layouts WMMA wants.
// ---------------------------------------------------------------------------

// x: [N,55] f32 row-major -> xpad: [N,64] f16 row-major (zero padded K)
__global__ void prep_x(const float* __restrict__ x, _Float16* __restrict__ xpad, int n) {
    int idx = blockIdx.x * blockDim.x + threadIdx.x;
    if (idx >= n * 64) return;
    int row = idx >> 6;
    int k   = idx & 63;
    xpad[idx] = (k < 55) ? (_Float16)x[row * 55 + k] : (_Float16)0.0f;
}

// W_fc: [55,128] f32 -> wt: [128,64] f16 (transposed, K-contiguous, padded)
__global__ void prep_wfc(const float* __restrict__ w, _Float16* __restrict__ wt) {
    int idx = blockIdx.x * blockDim.x + threadIdx.x;
    if (idx >= 128 * 64) return;
    int ncol = idx >> 6;
    int k    = idx & 63;
    wt[idx] = (k < 55) ? (_Float16)w[k * 128 + ncol] : (_Float16)0.0f;
}

// W: [L,128,128] f32 (k-major within layer) -> wt: [L,128,128] f16 transposed per layer
__global__ void prep_w(const float* __restrict__ w, _Float16* __restrict__ wt, int count) {
    int idx = blockIdx.x * blockDim.x + threadIdx.x;
    if (idx >= count) return;
    int layer = idx / (128 * 128);
    int r     = idx % (128 * 128);
    int ncol  = r >> 7;
    int k     = r & 127;
    wt[idx] = (_Float16)w[layer * 128 * 128 + k * 128 + ncol];
}

__global__ void zero_f32(float* __restrict__ p, int n) {
    int i = blockIdx.x * blockDim.x + threadIdx.x;
    if (i < n) p[i] = 0.0f;
}

// ---------------------------------------------------------------------------
// WMMA GEMM: Out[M,128] (f16) = act(A[M,K] (+ Aadd) @ Bt^T + bias)
//   A   : [nrows,K] f16 row-major (activation)
//   Aadd: [nrows,K] f32 row-major (optional neighbor aggregate), fused add
//   Bt  : [128,K]  f16, i.e. W transposed so each output col's K is contiguous
// One block = 256 threads = 8 waves; block owns 16 rows, wave w owns cols
// 16w..16w+15. K looped in 32-chunks with v_wmma_f32_16x16x32_f16.
// ---------------------------------------------------------------------------
template <int K, bool RELU, bool ADD>
__global__ __launch_bounds__(256) void gin_gemm(const _Float16* __restrict__ A,
                                                const float* __restrict__ Aadd,
                                                const _Float16* __restrict__ Bt,
                                                const float* __restrict__ bias,
                                                _Float16* __restrict__ Out,
                                                int nrows) {
    const int wave  = threadIdx.x >> 5;   // 0..7 : output column tile
    const int lane  = threadIdx.x & 31;
    const int m0    = blockIdx.x * 16;
    const int rlane = lane & 15;
    const int khalf = lane >> 4;          // 0/1 selects K-half per ISA layout

    int arow = m0 + rlane;
    if (arow >= nrows) arow = nrows - 1;  // clamp (stores are masked) — keeps EXEC full
    const int ncol = wave * 16 + rlane;

    const _Float16* aBase = A  + (size_t)arow * K + khalf * 8;
    const float*    gBase = ADD ? (Aadd + (size_t)arow * K + khalf * 8) : (const float*)0;
    const _Float16* bBase = Bt + (size_t)ncol * K + khalf * 8;

    v8f acc = {};
#pragma unroll
    for (int kt = 0; kt < K; kt += 32) {
        v16h a, b;
        // A fragment: elements 0..7 <- K = kt+8*khalf+0..7 ; 8..15 <- +16
        const v8h alo = *(const v8h*)(aBase + kt);
        const v8h ahi = *(const v8h*)(aBase + kt + 16);
        if (ADD) {
            const v8f glo = *(const v8f*)(gBase + kt);
            const v8f ghi = *(const v8f*)(gBase + kt + 16);
#pragma unroll
            for (int i = 0; i < 8; ++i) {
                a[i]     = (_Float16)((float)alo[i] + glo[i]);
                a[8 + i] = (_Float16)((float)ahi[i] + ghi[i]);
            }
        } else {
#pragma unroll
            for (int i = 0; i < 8; ++i) { a[i] = alo[i]; a[8 + i] = ahi[i]; }
        }
        // B fragment: elements 0..7 <- K = kt+8*khalf+0..7 ; 8..15 <- +16
        const v8h b0 = *(const v8h*)(bBase + kt);
        const v8h b1 = *(const v8h*)(bBase + kt + 16);
#pragma unroll
        for (int i = 0; i < 8; ++i) { b[i] = b0[i]; b[8 + i] = b1[i]; }

        acc = __builtin_amdgcn_wmma_f32_16x16x32_f16(false, a, false, b,
                                                     (short)0, acc, false, false);
    }

    const float bv = bias[ncol];
    _Float16* outp = Out + (size_t)(m0 + khalf * 8) * 128 + ncol;

    if (m0 + 16 <= nrows) {
        // Fast path (uniform branch): full tile, unguarded stores -> clean clause
#pragma unroll
        for (int r = 0; r < 8; ++r) {
            float v = acc[r] + bv;
            if (RELU) v = (v > 0.0f) ? v : 0.0f;
            outp[(size_t)r * 128] = (_Float16)v;   // C/D layout: M = r + 8*lane[4]
        }
    } else {
        // Ragged final tile: per-row guard
#pragma unroll
        for (int r = 0; r < 8; ++r) {
            float v = acc[r] + bv;
            if (RELU) v = (v > 0.0f) ? v : 0.0f;
            const int orow = m0 + r + khalf * 8;
            if (orow < nrows) Out[(size_t)orow * 128 + ncol] = (_Float16)v;
        }
    }
}

// ---------------------------------------------------------------------------
// Edge scatter: agg[dst] += h[src]. One wave per edge, 4 feats/lane.
// h is 25.6MB -> random gathers/atomics mostly L2-resident (192MB L2).
// ---------------------------------------------------------------------------
__global__ void gin_scatter(const _Float16* __restrict__ h,
                            const int* __restrict__ src,
                            const int* __restrict__ dst,
                            float* __restrict__ agg, int E) {
    int t = blockIdx.x * blockDim.x + threadIdx.x;
    int e = t >> 5;
    if (e >= E) return;
    int lane = t & 31;
    int s = src[e];
    int d = dst[e];
    const v4h hv = *(const v4h*)(h + (size_t)s * 128 + lane * 4);
    float* ap = agg + (size_t)d * 128 + lane * 4;
    atomicAdd(ap + 0, (float)hv[0]);
    atomicAdd(ap + 1, (float)hv[1]);
    atomicAdd(ap + 2, (float)hv[2]);
    atomicAdd(ap + 3, (float)hv[3]);
}

// Per-graph pooling: out[graph_ids[n]] += h[n]. One wave per node.
__global__ void gin_pool(const _Float16* __restrict__ h,
                         const int* __restrict__ gid,
                         float* __restrict__ out, int n) {
    int t = blockIdx.x * blockDim.x + threadIdx.x;
    int node = t >> 5;
    if (node >= n) return;
    int lane = t & 31;
    int g = gid[node];
    const v4h hv = *(const v4h*)(h + (size_t)node * 128 + lane * 4);
    float* op = out + (size_t)g * 128 + lane * 4;
    atomicAdd(op + 0, (float)hv[0]);
    atomicAdd(op + 1, (float)hv[1]);
    atomicAdd(op + 2, (float)hv[2]);
    atomicAdd(op + 3, (float)hv[3]);
}

// ---------------------------------------------------------------------------

extern "C" void kernel_launch(void* const* d_in, const int* in_sizes, int n_in,
                              void* d_out, int out_size, void* d_ws, size_t ws_size,
                              hipStream_t stream) {
    const float* x   = (const float*)d_in[0];
    const float* Wfc = (const float*)d_in[1];
    const float* bfc = (const float*)d_in[2];
    const float* W1  = (const float*)d_in[3];
    const float* b1  = (const float*)d_in[4];
    const float* W2  = (const float*)d_in[5];
    const float* b2  = (const float*)d_in[6];
    const int*   src = (const int*)d_in[7];
    const int*   dst = (const int*)d_in[8];
    const int*   gid = (const int*)d_in[9];

    const int N = in_sizes[0] / 55;          // 100000
    const int E = in_sizes[7];               // 1600000
    const int Gn = out_size / 128;           // 4096
    const int L = in_sizes[4] / 128;         // 3

    // Workspace carve-up (256B aligned)
    char* ws = (char*)d_ws;
    size_t off = 0;
    auto carve = [&](size_t bytes) {
        char* p = ws + off;
        off = (off + bytes + 255) & ~(size_t)255;
        return p;
    };
    _Float16* xpad = (_Float16*)carve((size_t)N * 64 * sizeof(_Float16));
    _Float16* h16  = (_Float16*)carve((size_t)N * 128 * sizeof(_Float16));
    _Float16* z16  = (_Float16*)carve((size_t)N * 128 * sizeof(_Float16));
    float*    agg  = (float*)   carve((size_t)N * 128 * sizeof(float));
    _Float16* wfct = (_Float16*)carve((size_t)128 * 64 * sizeof(_Float16));
    _Float16* w1t  = (_Float16*)carve((size_t)L * 128 * 128 * sizeof(_Float16));
    _Float16* w2t  = (_Float16*)carve((size_t)L * 128 * 128 * sizeof(_Float16));
    (void)ws_size;

    const int TPB = 256;
    const int mtiles = (N + 15) / 16;

    // --- prep ---
    prep_x  <<<(N * 64 + TPB - 1) / TPB, TPB, 0, stream>>>(x, xpad, N);
    prep_wfc<<<(128 * 64 + TPB - 1) / TPB, TPB, 0, stream>>>(Wfc, wfct);
    prep_w  <<<(L * 128 * 128 + TPB - 1) / TPB, TPB, 0, stream>>>(W1, w1t, L * 128 * 128);
    prep_w  <<<(L * 128 * 128 + TPB - 1) / TPB, TPB, 0, stream>>>(W2, w2t, L * 128 * 128);

    // --- node projection: h = x @ W_fc + b_fc ---
    gin_gemm<64, false, false><<<mtiles, TPB, 0, stream>>>(
        xpad, (const float*)0, wfct, bfc, h16, N);

    // --- L x GINConv ---
    for (int i = 0; i < L; ++i) {
        zero_f32<<<(N * 128 + TPB - 1) / TPB, TPB, 0, stream>>>(agg, N * 128);
        gin_scatter<<<((size_t)E * 32 + TPB - 1) / TPB, TPB, 0, stream>>>(
            h16, src, dst, agg, E);
        // z = relu((h + agg) @ W1[i] + b1[i])
        gin_gemm<128, true, true><<<mtiles, TPB, 0, stream>>>(
            h16, agg, w1t + (size_t)i * 128 * 128, b1 + i * 128, z16, N);
        // h = z @ W2[i] + b2[i]
        gin_gemm<128, false, false><<<mtiles, TPB, 0, stream>>>(
            z16, (const float*)0, w2t + (size_t)i * 128 * 128, b2 + i * 128, h16, N);
    }

    // --- per-graph sum pooling ---
    zero_f32<<<(Gn * 128 + TPB - 1) / TPB, TPB, 0, stream>>>((float*)d_out, Gn * 128);
    gin_pool<<<((size_t)N * 32 + TPB - 1) / TPB, TPB, 0, stream>>>(
        h16, gid, (float*)d_out, N);
}
